// GraphAttentionTransformerMD17_57294863729171
// MI455X (gfx1250) — compile-verified, matrix operand order checked by reading, not run on hardware
//
#include <hip/hip_runtime.h>
#include <hip/hip_bf16.h>

typedef __attribute__((ext_vector_type(16))) __bf16 v16bf;
typedef __attribute__((ext_vector_type(8)))  float  v8f;

#define N_NODES 30000
#define N_EDGES 480000
#define NBATCH  64

// ---- helpers ---------------------------------------------------------------

union FragBF { v16bf v; uint4 u[2]; };

__device__ __forceinline__ unsigned short f32_bf16(float f) {
    unsigned int u = __float_as_uint(f);
    u += 0x7FFFu + ((u >> 16) & 1u);          // round-to-nearest-even
    return (unsigned short)(u >> 16);
}
__device__ __forceinline__ float bf16_f32(unsigned short h) {
    return __uint_as_float(((unsigned int)h) << 16);
}
__device__ __forceinline__ float fast_rcp(float x) { return __builtin_amdgcn_rcpf(x); }
__device__ __forceinline__ float siluf(float x) { return x * fast_rcp(1.0f + __expf(-x)); }
__device__ __forceinline__ float sigmf(float x) { return fast_rcp(1.0f + __expf(-x)); }

// Load a 16x32 bf16 fragment from a row-major [rows][stride] bf16 buffer.
// CDNA5 16-bit A layout: lanes 0-15 -> row r, K 0-7 & 16-23; lanes 16-31 -> K 8-15 & 24-31.
// B fragments come from pre-transposed weights W^T [N][K], identical pattern with row = n.
__device__ __forceinline__ v16bf load_frag(const unsigned short* base, int stride, int lane) {
    const int hh = (lane >> 4) & 1;
    const int r  = lane & 15;
    const unsigned short* p = base + r * stride + hh * 8;
    FragBF f;
    f.u[0] = *reinterpret_cast<const uint4*>(p);        // K 0-7  / 8-15
    f.u[1] = *reinterpret_cast<const uint4*>(p + 16);   // K 16-23 / 24-31
    return f.v;
}

__device__ __forceinline__ v8f wmma_step(v16bf a, v16bf b, v8f c) {
    return __builtin_amdgcn_wmma_f32_16x16x32_bf16(false, a, false, b, (short)0, c, false, false);
}

// ---- weight transpose + bf16 convert --------------------------------------

__global__ void transpose_bf16_kernel(const float* __restrict__ src,
                                      unsigned short* __restrict__ dst,
                                      int K, int Nout) {
    int i = blockIdx.x * 256 + threadIdx.x;
    if (i < K * Nout) {
        int k = i / Nout, n = i - k * Nout;   // src[k][n] -> dst[n][k]
        dst[n * K + k] = f32_bf16(src[i]);
    }
}

// ---- edge kernel: geometry + RBF + MLP + projections + scatter -------------

__global__ __launch_bounds__(128) void edge_kernel(
    const float* __restrict__ node_feat, const float* __restrict__ pos,
    const int* __restrict__ edge_index,
    const float* __restrict__ b1, const float* __restrict__ b2, const float* __restrict__ b3,
    const unsigned short* __restrict__ w1T, const unsigned short* __restrict__ w2T,
    const unsigned short* __restrict__ w3T, const unsigned short* __restrict__ lin0T,
    const unsigned short* __restrict__ lin1T, const unsigned short* __restrict__ lin2T,
    float* __restrict__ nodeOut)
{
    __shared__ __align__(16) unsigned short sA[4][16 * 128];
    __shared__ __align__(16) unsigned short sB[4][16 * 192];
    __shared__ int   sSrc[4][16];
    __shared__ int   sDst[4][16];
    __shared__ float sDist[4][16];
    __shared__ float sCut[4][16];
    __shared__ float sSh1[4][16][3];
    __shared__ float sSh2[4][16][5];

    const int lane = threadIdx.x & 31;
    const int wv   = threadIdx.x >> 5;
    const int e0   = blockIdx.x * 64 + wv * 16;
    unsigned short* bufA = sA[wv];
    unsigned short* bufB = sB[wv];

    // ---- geometry + spherical harmonics (one edge per lane, lanes 0-15) ----
    if (lane < 16) {
        int e = e0 + lane;
        int si = edge_index[e];
        int di = edge_index[N_EDGES + e];
        float rx = pos[di * 3 + 0] - pos[si * 3 + 0];
        float ry = pos[di * 3 + 1] - pos[si * 3 + 1];
        float rz = pos[di * 3 + 2] - pos[si * 3 + 2];
        float dist = sqrtf(rx * rx + ry * ry + rz * rz + 1e-12f);
        float inv = fast_rcp(dist);
        float ux = rx * inv, uy = ry * inv, uz = rz * inv;
        float cut = dist < 5.0f ? 0.5f * (__cosf(dist * 0.62831853071795865f) + 1.0f) : 0.0f;
        const float s3 = 1.7320508075688772f;
        const float s5 = 2.2360679774997896f;
        const float s15 = 3.8729833462074170f;
        sSrc[wv][lane] = si; sDst[wv][lane] = di;
        sDist[wv][lane] = dist; sCut[wv][lane] = cut;
        sSh1[wv][lane][0] = s3 * ux; sSh1[wv][lane][1] = s3 * uy; sSh1[wv][lane][2] = s3 * uz;
        sSh2[wv][lane][0] = s15 * ux * uy;
        sSh2[wv][lane][1] = s15 * uy * uz;
        sSh2[wv][lane][2] = 0.5f * s5 * (3.0f * uz * uz - 1.0f);
        sSh2[wv][lane][3] = s15 * uz * ux;
        sSh2[wv][lane][4] = 0.5f * s15 * (ux * ux - uy * uy);
    }
    __builtin_amdgcn_wave_barrier();

    // ---- RBF features -> bufA (16 x 128 bf16) ----
    const float rbf_start = 0.006737946999085467f;            // exp(-5)
    const float rbf_span  = 1.0f - rbf_start;
    const float rbf_beta  = 4151.76f;                          // (2/128*(1-start))^-2
    for (int i = lane; i < 16 * 128; i += 32) {
        int e = i >> 7, r = i & 127;
        float d = sDist[wv][e], c = sCut[wv][e];
        float mean = rbf_start + rbf_span * ((float)r * (1.0f / 127.0f));
        float t = __expf(-d) - mean;
        bufA[i] = f32_bf16(c * __expf(-rbf_beta * t * t));
    }
    __builtin_amdgcn_wave_barrier();

    const int hh  = lane >> 4;
    const int col = lane & 15;

    // ---- L1: h = silu(rbf @ w1 + b1)  (K=128, N=64) -> bufB (stride 192) ----
    for (int nt = 0; nt < 4; ++nt) {
        v8f acc = {};
        for (int ks = 0; ks < 4; ++ks)
            acc = wmma_step(load_frag(bufA + ks * 32, 128, lane),
                            load_frag(w1T + (nt * 16) * 128 + ks * 32, 128, lane), acc);
        int ch = nt * 16 + col; float bias = b1[ch];
        for (int v = 0; v < 8; ++v)
            bufB[(v + 8 * hh) * 192 + ch] = f32_bf16(siluf(acc[v] + bias));
    }
    __builtin_amdgcn_wave_barrier();

    // ---- L2: h2 = silu(h @ w2 + b2)  (K=64, N=64) -> bufA (stride 128) ----
    for (int nt = 0; nt < 4; ++nt) {
        v8f acc = {};
        for (int ks = 0; ks < 2; ++ks)
            acc = wmma_step(load_frag(bufB + ks * 32, 192, lane),
                            load_frag(w2T + (nt * 16) * 64 + ks * 32, 64, lane), acc);
        int ch = nt * 16 + col; float bias = b2[ch];
        for (int v = 0; v < 8; ++v)
            bufA[(v + 8 * hh) * 128 + ch] = f32_bf16(siluf(acc[v] + bias));
    }
    __builtin_amdgcn_wave_barrier();

    // ---- L3: tw = h2 @ w3 + b3  (K=64, N=192) -> bufB (stride 192) ----
    for (int nt = 0; nt < 12; ++nt) {
        v8f acc = {};
        for (int ks = 0; ks < 2; ++ks)
            acc = wmma_step(load_frag(bufA + ks * 32, 128, lane),
                            load_frag(w3T + (nt * 16) * 64 + ks * 32, 64, lane), acc);
        int ch = nt * 16 + col; float bias = b3[ch];
        for (int v = 0; v < 8; ++v)
            bufB[(v + 8 * hh) * 192 + ch] = f32_bf16(acc[v] + bias);
    }
    __builtin_amdgcn_wave_barrier();

    // ---- gather x = node_feat[src]; m0 = x*tw0 -> bufA[0:64], xw1 -> bufA[64:128] ----
    for (int i = lane; i < 16 * 64; i += 32) {
        int e = i >> 6, c = i & 63;
        float x = node_feat[sSrc[wv][e] * 64 + c];
        bufA[e * 128 + c]      = f32_bf16(x * bf16_f32(bufB[e * 192 + c]));
        bufA[e * 128 + 64 + c] = f32_bf16(x * bf16_f32(bufB[e * 192 + 64 + c]));
    }
    __builtin_amdgcn_wave_barrier();
    // xw2 -> bufB[0:64] (tw0 fully consumed above; reads tw2 from cols 128..191)
    for (int i = lane; i < 16 * 64; i += 32) {
        int e = i >> 6, c = i & 63;
        float x = node_feat[sSrc[wv][e] * 64 + c];
        bufB[e * 192 + c] = f32_bf16(x * bf16_f32(bufB[e * 192 + 128 + c]));
    }
    __builtin_amdgcn_wave_barrier();

    const float inv_deg = 0.25335270f;                 // 1/sqrt(AVG_DEGREE)

    // ---- gates g = sigmoid(s[:,128:224]) : s-tiles 8..13, kept in registers ----
    float gv[6][8];
    for (int j = 0; j < 6; ++j) {
        int nt = 8 + j;
        v8f acc = {};
        for (int ks = 0; ks < 2; ++ks)
            acc = wmma_step(load_frag(bufA + ks * 32, 128, lane),
                            load_frag(lin0T + (nt * 16) * 64 + ks * 32, 64, lane), acc);
        for (int v = 0; v < 8; ++v) gv[j][v] = sigmf(acc[v]);
    }

    // ---- scal = silu(s[:,:128]) -> scatter into n0 ----
    for (int nt = 0; nt < 8; ++nt) {
        v8f acc = {};
        for (int ks = 0; ks < 2; ++ks)
            acc = wmma_step(load_frag(bufA + ks * 32, 128, lane),
                            load_frag(lin0T + (nt * 16) * 64 + ks * 32, 64, lane), acc);
        for (int v = 0; v < 8; ++v) {
            int e = v + 8 * hh;
            int dn = sDst[wv][e];
            atomicAdd(nodeOut + dn * 480 + nt * 16 + col, siluf(acc[v]) * inv_deg);
        }
    }

    // ---- y1 = (x*tw1)@lin1, gated, outer sh1 -> scatter into n1 ----
    for (int nt = 0; nt < 4; ++nt) {
        v8f acc = {};
        for (int ks = 0; ks < 2; ++ks)
            acc = wmma_step(load_frag(bufA + 64 + ks * 32, 128, lane),
                            load_frag(lin1T + (nt * 16) * 64 + ks * 32, 64, lane), acc);
        for (int v = 0; v < 8; ++v) {
            int e = v + 8 * hh;
            int dn = sDst[wv][e];
            float val = acc[v] * gv[nt][v] * inv_deg;
            float* p = nodeOut + dn * 480 + 128 + (nt * 16 + col) * 3;
            atomicAdd(p + 0, val * sSh1[wv][e][0]);
            atomicAdd(p + 1, val * sSh1[wv][e][1]);
            atomicAdd(p + 2, val * sSh1[wv][e][2]);
        }
    }

    // ---- y2 = (x*tw2)@lin2, gated, outer sh2 -> scatter into n2 ----
    for (int nt = 0; nt < 2; ++nt) {
        v8f acc = {};
        for (int ks = 0; ks < 2; ++ks)
            acc = wmma_step(load_frag(bufB + ks * 32, 192, lane),
                            load_frag(lin2T + (nt * 16) * 64 + ks * 32, 64, lane), acc);
        for (int v = 0; v < 8; ++v) {
            int e = v + 8 * hh;
            int dn = sDst[wv][e];
            float val = acc[v] * gv[4 + nt][v] * inv_deg;
            float* p = nodeOut + dn * 480 + 320 + (nt * 16 + col) * 5;
            for (int j = 0; j < 5; ++j) atomicAdd(p + j, val * sSh2[wv][e][j]);
        }
    }
}

// ---- node kernel: fc_inv GEMM + LN + head MLP + energy scatter -------------

__global__ __launch_bounds__(32) void node_kernel(
    const float* __restrict__ nodeOut,
    const unsigned short* __restrict__ fcT, const unsigned short* __restrict__ hw1T,
    const float* __restrict__ ln_g, const float* __restrict__ ln_b,
    const float* __restrict__ hb1, const float* __restrict__ hw2,
    const float* __restrict__ hb2,
    const int* __restrict__ batch, float* __restrict__ energy)
{
    __shared__ __align__(16) unsigned short sN0[16 * 128];
    __shared__ float sF[16 * 512];
    __shared__ __align__(16) unsigned short sFH[16 * 512];
    __shared__ float sMu[16], sRstd[16];

    const int lane = threadIdx.x & 31;
    const int t0 = blockIdx.x * 16;
    const int hh = lane >> 4, col = lane & 15;

    for (int i = lane; i < 16 * 128; i += 32) {
        int r = i >> 7, c = i & 127;
        sN0[i] = f32_bf16(nodeOut[(t0 + r) * 480 + c]);
    }
    __builtin_amdgcn_wave_barrier();

    // f = n0 @ fc_inv (K=128, N=512) -> sF
    for (int nt = 0; nt < 32; ++nt) {
        v8f acc = {};
        for (int ks = 0; ks < 4; ++ks)
            acc = wmma_step(load_frag(sN0 + ks * 32, 128, lane),
                            load_frag(fcT + (nt * 16) * 128 + ks * 32, 128, lane), acc);
        int ch = nt * 16 + col;
        for (int v = 0; v < 8; ++v) sF[(v + 8 * hh) * 512 + ch] = acc[v];
    }
    __builtin_amdgcn_wave_barrier();

    // layernorm: lanes (r, r+16) cooperate on row r
    {
        int r = col;
        float s = 0.0f;
        for (int c = hh * 256; c < hh * 256 + 256; ++c) s += sF[r * 512 + c];
        s += __shfl_xor(s, 16, 32);
        float mu = s * (1.0f / 512.0f);
        float vr = 0.0f;
        for (int c = hh * 256; c < hh * 256 + 256; ++c) {
            float d = sF[r * 512 + c] - mu; vr += d * d;
        }
        vr += __shfl_xor(vr, 16, 32);
        float rstd = rsqrtf(vr * (1.0f / 512.0f) + 1e-5f);
        if (hh == 0) { sMu[r] = mu; sRstd[r] = rstd; }
    }
    __builtin_amdgcn_wave_barrier();
    for (int i = lane; i < 16 * 512; i += 32) {
        int r = i >> 9, c = i & 511;
        sFH[i] = f32_bf16((sF[i] - sMu[r]) * sRstd[r] * ln_g[c] + ln_b[c]);
    }
    __builtin_amdgcn_wave_barrier();

    // head: out = silu(fh @ hw1 + hb1) @ hw2 + hb2 (fold hw2 into tile loop)
    float part[8] = {0, 0, 0, 0, 0, 0, 0, 0};
    for (int nt = 0; nt < 32; ++nt) {
        v8f acc = {};
        for (int ks = 0; ks < 16; ++ks)
            acc = wmma_step(load_frag(sFH + ks * 32, 512, lane),
                            load_frag(hw1T + (nt * 16) * 512 + ks * 32, 512, lane), acc);
        int ch = nt * 16 + col;
        float b = hb1[ch], w = hw2[ch];
        for (int v = 0; v < 8; ++v) part[v] += siluf(acc[v] + b) * w;
    }
    // reduce across the 16 lane-columns within each half-wave
    for (int off = 1; off < 16; off <<= 1)
        for (int v = 0; v < 8; ++v) part[v] += __shfl_xor(part[v], off, 32);

    if (col == 0) {
        const float inv_nodes = 0.23550291f;          // 1/sqrt(AVG_NUM_NODES)
        float b2 = hb2[0];
        for (int v = 0; v < 8; ++v) {
            int node = t0 + v + 8 * hh;
            atomicAdd(energy + batch[node], (part[v] + b2) * inv_nodes);
        }
    }
}

// ---- launch ----------------------------------------------------------------

extern "C" void kernel_launch(void* const* d_in, const int* in_sizes, int n_in,
                              void* d_out, int out_size, void* d_ws, size_t ws_size,
                              hipStream_t stream) {
    const float* node_feat = (const float*)d_in[0];
    const float* pos       = (const float*)d_in[1];
    const float* w1        = (const float*)d_in[2];
    const float* b1        = (const float*)d_in[3];
    const float* w2        = (const float*)d_in[4];
    const float* b2        = (const float*)d_in[5];
    const float* w3        = (const float*)d_in[6];
    const float* b3        = (const float*)d_in[7];
    const float* lin0      = (const float*)d_in[8];
    const float* lin1      = (const float*)d_in[9];
    const float* lin2      = (const float*)d_in[10];
    const float* fc_inv    = (const float*)d_in[11];
    const float* ln_g      = (const float*)d_in[12];
    const float* ln_b      = (const float*)d_in[13];
    const float* hw1       = (const float*)d_in[14];
    const float* hb1       = (const float*)d_in[15];
    const float* hw2       = (const float*)d_in[16];
    const float* hb2       = (const float*)d_in[17];
    const int* edge_index  = (const int*)d_in[18];
    const int* batch       = (const int*)d_in[19];

    float* out = (float*)d_out;
    hipMemsetAsync(d_out, 0, (size_t)out_size * sizeof(float), stream);

    // bf16 transposed weights in workspace (all 16B-aligned offsets)
    unsigned short* ws  = (unsigned short*)d_ws;
    unsigned short* w1T = ws;                 // 64  x 128
    unsigned short* w2T = w1T + 64 * 128;     // 64  x 64
    unsigned short* w3T = w2T + 64 * 64;      // 192 x 64
    unsigned short* l0T = w3T + 192 * 64;     // 224 x 64
    unsigned short* l1T = l0T + 224 * 64;     // 64  x 64
    unsigned short* l2T = l1T + 64 * 64;      // 32  x 64
    unsigned short* fcT = l2T + 32 * 64;      // 512 x 128
    unsigned short* h1T = fcT + 512 * 128;    // 512 x 512

    auto tr = [&](const float* s, unsigned short* d, int K, int Nout) {
        int tot = K * Nout;
        transpose_bf16_kernel<<<(tot + 255) / 256, 256, 0, stream>>>(s, d, K, Nout);
    };
    tr(w1,     w1T, 128, 64);
    tr(w2,     w2T, 64, 64);
    tr(w3,     w3T, 64, 192);
    tr(lin0,   l0T, 64, 224);
    tr(lin1,   l1T, 64, 64);
    tr(lin2,   l2T, 64, 32);
    tr(fc_inv, fcT, 128, 512);
    tr(hw1,    h1T, 512, 512);

    float* nodeOut = out + NBATCH;  // node_out region, after energy[64]

    edge_kernel<<<N_EDGES / 64, 128, 0, stream>>>(
        node_feat, pos, edge_index, b1, b2, b3,
        w1T, w2T, w3T, l0T, l1T, l2T, nodeOut);

    node_kernel<<<N_NODES / 16, 32, 0, stream>>>(
        nodeOut, fcT, h1T, ln_g, ln_b, hb1, hw2, hb2, batch, out);
}
